// MPNEncoder_50182397887184
// MI455X (gfx1250) — matrix operand order repaired and checked by reading, not compiled, not run on hardware
//
#include <hip/hip_runtime.h>

#define N_ATOMS   100000
#define N_BONDS   200000
#define MAX_NB    6
#define ATOM_FDIM 133
#define BOND_FDIM 147
#define HIDDEN    512
#define N_MOLS    8192

#define KB_BONDS  160   // 147 padded to multiple of 32
#define KB_CAT    672   // 645 padded to multiple of 32

typedef __attribute__((ext_vector_type(16))) __bf16 v16bf;
typedef __attribute__((ext_vector_type(8)))  float  v8f;

__device__ __forceinline__ unsigned short f2bf(float f) {
  unsigned u = __float_as_uint(f);
  u += 0x7FFFu + ((u >> 16) & 1u);          // round-to-nearest-even
  return (unsigned short)(u >> 16);
}
__device__ __forceinline__ float bf2f(unsigned short h) {
  return __uint_as_float(((unsigned)h) << 16);
}
__device__ __forceinline__ unsigned pack2bf(float a, float b) {
  return (unsigned)f2bf(a) | ((unsigned)f2bf(b) << 16);
}

// ---------------------------------------------------------------- utilities
__global__ void zero_f32(float* __restrict__ p, int n) {
  int i = blockIdx.x * blockDim.x + threadIdx.x;
  if (i < n) p[i] = 0.0f;
}

// Wt[n*Kpad + k] = (k < K) ? W[k*N + n] : 0   (bf16, transposed to N-major)
__global__ void conv_transpose(const float* __restrict__ W,
                               unsigned short* __restrict__ Wt,
                               int K, int N, int Kpad) {
  int idx = blockIdx.x * blockDim.x + threadIdx.x;
  if (idx >= N * Kpad) return;
  int n = idx / Kpad, k = idx - n * Kpad;
  Wt[idx] = (k < K) ? f2bf(W[(size_t)k * N + n]) : (unsigned short)0;
}

// Y[r*Kpad + k] = (k < K) ? bf16(X[r*K + k]) : 0
__global__ void conv_pad(const float* __restrict__ X,
                         unsigned short* __restrict__ Y,
                         int rows, int K, int Kpad) {
  int idx = blockIdx.x * blockDim.x + threadIdx.x;
  if (idx >= rows * Kpad) return;
  int r = idx / Kpad, k = idx - r * Kpad;
  Y[idx] = (k < K) ? f2bf(X[(size_t)r * K + k]) : (unsigned short)0;
}

// ---------------------------------------------------------------- WMMA GEMM
// C[M,512] = A[M,K](bf16,row-major) * Bt[512,K](bf16, N-major == W^T)
// MODE 0: inp = C (f32);  msg = bf16(relu(C))
// MODE 1: msg = bf16(relu(inp + C))
// MODE 2: h = relu(C + b_o[n]); atomicAdd(sums[a2m[m]][n], h)
// FULL  : all BM rows of this block are in range (no guards anywhere)
#define BM 128
#define BN 64
#define BK 32
#define LDT 40   // padded LDS row stride in halfs (80B, 16B aligned)

template<int MODE, bool FULL>
__global__ __launch_bounds__(256) void wmma_gemm(
    const unsigned short* __restrict__ A, int mbase, int M, int K,
    const unsigned short* __restrict__ Bt,
    float* __restrict__ inp,
    unsigned short* __restrict__ msg,
    const float* __restrict__ bo,
    const int* __restrict__ a2m,
    float* __restrict__ sums) {
  __shared__ __align__(16) unsigned short sA[BM * LDT];
  __shared__ __align__(16) unsigned short sB[BN * LDT];

  const int tid  = threadIdx.x;
  const int lane = tid & 31;
  const int wave = tid >> 5;
  const int wm = wave & 3;   // 32-row group
  const int wn = wave >> 2;  // 32-col group
  const int m0 = mbase + blockIdx.y * BM;
  const int n0 = blockIdx.x * BN;

  const int half = lane >> 4;   // 0/1 : lane-half select for WMMA layouts
  const int l16  = lane & 15;

  v8f acc[2][2] = {};

  const int arow = tid >> 1;          // 0..127
  const int acol = (tid & 1) * 16;    // 0 / 16
  const int brow = tid >> 2;          // 0..63
  const int bcol = (tid & 3) * 8;     // 0/8/16/24

  const int gr = m0 + arow;
  const unsigned short* aptr = A + (size_t)gr * K + acol;   // + kt each step
  const unsigned short* bptr = Bt + (size_t)(n0 + brow) * K + bcol;
  const bool arow_ok = FULL || (gr < M);

  union Fr { v16bf bf; uint4 u[2]; };

  for (int kt = 0; kt < K; kt += BK) {
    // stage A tile (BM x BK): 16 halfs / thread
    {
      uint4 d0 = {0u, 0u, 0u, 0u}, d1 = d0;
      if (arow_ok) {
        const uint4* p = (const uint4*)(aptr + kt);
        d0 = p[0];
        d1 = p[1];
        if (FULL) __builtin_prefetch(aptr + kt + 2 * BK, 0, 1);
      }
      *(uint4*)(sA + arow * LDT + acol)     = d0;
      *(uint4*)(sA + arow * LDT + acol + 8) = d1;
    }
    // stage B tile (BN x BK): 8 halfs / thread (Bt is N-major, contiguous K)
    *(uint4*)(sB + brow * LDT + bcol) = *(const uint4*)(bptr + kt);
    __syncthreads();

    // A fragment: lane-half h holds K-pairs {8h..8h+7} and {8h+16..8h+23}
    Fr a[2], b[2];
    for (int fm = 0; fm < 2; ++fm) {
      const unsigned short* pr = sA + (wm * 32 + fm * 16 + l16) * LDT + half * 8;
      a[fm].u[0] = *(const uint4*)(pr);
      a[fm].u[1] = *(const uint4*)(pr + 16);
    }
    // B fragment: lane-half h holds K {8h..8h+7} and {8h+16..8h+23}, column = l16
    for (int fn = 0; fn < 2; ++fn) {
      const unsigned short* pr = sB + (wn * 32 + fn * 16 + l16) * LDT + half * 8;
      b[fn].u[0] = *(const uint4*)(pr);
      b[fn].u[1] = *(const uint4*)(pr + 16);
    }
    for (int fm = 0; fm < 2; ++fm)
      for (int fn = 0; fn < 2; ++fn)
        acc[fm][fn] = __builtin_amdgcn_wmma_f32_16x16x32_bf16(
            false, a[fm].bf, false, b[fn].bf, (short)0, acc[fm][fn],
            false, false);
    __syncthreads();
  }

  // epilogue: C layout = VGPR r, lane<16 -> M=r, lane>=16 -> M=8+r, N=l16
  for (int fm = 0; fm < 2; ++fm)
    for (int fn = 0; fn < 2; ++fn) {
      const int n = n0 + wn * 32 + fn * 16 + l16;
      for (int r = 0; r < 8; ++r) {
        const int m = m0 + wm * 32 + fm * 16 + half * 8 + r;
        if (FULL || m < M) {
          float v = acc[fm][fn][r];
          if (MODE == 0) {
            inp[(size_t)m * HIDDEN + n] = v;
            msg[(size_t)m * HIDDEN + n] = f2bf(v > 0.0f ? v : 0.0f);
          } else if (MODE == 1) {
            float s = inp[(size_t)m * HIDDEN + n] + v;
            msg[(size_t)m * HIDDEN + n] = f2bf(s > 0.0f ? s : 0.0f);
          } else {
            float h = v + bo[n];
            h = h > 0.0f ? h : 0.0f;
            atomicAdd(&sums[(size_t)a2m[m] * HIDDEN + n], h);
          }
        }
      }
    }
}

// ------------------------------------------------------- gather / scatter
// amsg[a] = sum_j msg[a2b[a][j]]  (f32 accumulate from bf16)
// 4 atoms per 256-thread block, 64 lanes x 8 cols (b128 traffic)
__global__ __launch_bounds__(256) void gather_atoms(
    const unsigned short* __restrict__ msg,
    const int* __restrict__ a2b,
    float* __restrict__ amsg) {
  const int a = blockIdx.x * 4 + (threadIdx.x >> 6);
  const int c = (threadIdx.x & 63) * 8;
  float s0 = 0.f, s1 = 0.f, s2 = 0.f, s3 = 0.f, s4 = 0.f, s5 = 0.f, s6 = 0.f, s7 = 0.f;
  #pragma unroll
  for (int j = 0; j < MAX_NB; ++j) {
    const int b = a2b[a * MAX_NB + j];
    const uint4 u = *(const uint4*)(msg + (size_t)b * HIDDEN + c);
    s0 += bf2f((unsigned short)(u.x & 0xFFFFu)); s1 += bf2f((unsigned short)(u.x >> 16));
    s2 += bf2f((unsigned short)(u.y & 0xFFFFu)); s3 += bf2f((unsigned short)(u.y >> 16));
    s4 += bf2f((unsigned short)(u.z & 0xFFFFu)); s5 += bf2f((unsigned short)(u.z >> 16));
    s6 += bf2f((unsigned short)(u.w & 0xFFFFu)); s7 += bf2f((unsigned short)(u.w >> 16));
  }
  float4* o = (float4*)(amsg + (size_t)a * HIDDEN + c);
  o[0] = make_float4(s0, s1, s2, s3);
  o[1] = make_float4(s4, s5, s6, s7);
}

// t[b] = bf16( amsg[b2a[b]] - msg[b2revb[b]] ), 4 bonds / block, b128 traffic
__global__ __launch_bounds__(256) void bond_pre(
    const float* __restrict__ amsg,
    const unsigned short* __restrict__ msg,
    const int* __restrict__ b2a,
    const int* __restrict__ b2revb,
    unsigned short* __restrict__ t) {
  const int b = blockIdx.x * 4 + (threadIdx.x >> 6);
  const int c = (threadIdx.x & 63) * 8;
  const int a  = b2a[b];
  const int rb = b2revb[b];
  const uint4 u = *(const uint4*)(msg + (size_t)rb * HIDDEN + c);
  const float4* m = (const float4*)(amsg + (size_t)a * HIDDEN + c);
  const float4 m0 = m[0], m1 = m[1];
  uint4 o;
  o.x = pack2bf(m0.x - bf2f((unsigned short)(u.x & 0xFFFFu)),
                m0.y - bf2f((unsigned short)(u.x >> 16)));
  o.y = pack2bf(m0.z - bf2f((unsigned short)(u.y & 0xFFFFu)),
                m0.w - bf2f((unsigned short)(u.y >> 16)));
  o.z = pack2bf(m1.x - bf2f((unsigned short)(u.z & 0xFFFFu)),
                m1.y - bf2f((unsigned short)(u.z >> 16)));
  o.w = pack2bf(m1.z - bf2f((unsigned short)(u.w & 0xFFFFu)),
                m1.w - bf2f((unsigned short)(u.w >> 16)));
  *(uint4*)(t + (size_t)b * HIDDEN + c) = o;
}

// ain[a] = bf16( concat(f_atoms[a], sum_j msg[a2b[a][j]]) ) padded to 672
__global__ void build_a_input(const float* __restrict__ f_atoms,
                              const unsigned short* __restrict__ msg,
                              const int* __restrict__ a2b,
                              unsigned short* __restrict__ ain) {
  const int a = blockIdx.x;
  int nb[MAX_NB];
  #pragma unroll
  for (int j = 0; j < MAX_NB; ++j) nb[j] = a2b[a * MAX_NB + j];
  for (int c = threadIdx.x; c < KB_CAT; c += blockDim.x) {
    unsigned short o;
    if (c < ATOM_FDIM) {
      o = f2bf(f_atoms[(size_t)a * ATOM_FDIM + c]);
    } else if (c < ATOM_FDIM + HIDDEN) {
      const int h = c - ATOM_FDIM;
      float s = 0.0f;
      #pragma unroll
      for (int j = 0; j < MAX_NB; ++j)
        s += bf2f(msg[(size_t)nb[j] * HIDDEN + h]);
      o = f2bf(s);
    } else {
      o = 0;
    }
    ain[(size_t)a * KB_CAT + c] = o;
  }
}

__global__ void count_atoms(const int* __restrict__ a2m,
                            float* __restrict__ counts) {
  int a = blockIdx.x * blockDim.x + threadIdx.x;
  if (a < N_ATOMS) atomicAdd(&counts[a2m[a]], 1.0f);
}

__global__ void finalize_mean(float* __restrict__ out,
                              const float* __restrict__ counts) {
  int i = blockIdx.x * blockDim.x + threadIdx.x;
  if (i < N_MOLS * HIDDEN) {
    const float c = counts[i / HIDDEN];
    out[i] = (c > 0.0f) ? out[i] / c : 0.0f;
  }
}

// ---------------------------------------------------------------- launcher
static inline void launch_gemm(int mode, const unsigned short* A, int M, int K,
                               const unsigned short* Bt, float* inp,
                               unsigned short* msg, const float* bo,
                               const int* a2m, float* sums, hipStream_t stream) {
  const int fullRows = (M / BM) * BM;
  if (fullRows > 0) {
    dim3 g(HIDDEN / BN, M / BM);
    if (mode == 0)
      wmma_gemm<0, true><<<g, 256, 0, stream>>>(A, 0, M, K, Bt, inp, msg, bo, a2m, sums);
    else if (mode == 1)
      wmma_gemm<1, true><<<g, 256, 0, stream>>>(A, 0, M, K, Bt, inp, msg, bo, a2m, sums);
    else
      wmma_gemm<2, true><<<g, 256, 0, stream>>>(A, 0, M, K, Bt, inp, msg, bo, a2m, sums);
  }
  if (fullRows < M) {
    dim3 g(HIDDEN / BN, 1);
    if (mode == 0)
      wmma_gemm<0, false><<<g, 256, 0, stream>>>(A, fullRows, M, K, Bt, inp, msg, bo, a2m, sums);
    else if (mode == 1)
      wmma_gemm<1, false><<<g, 256, 0, stream>>>(A, fullRows, M, K, Bt, inp, msg, bo, a2m, sums);
    else
      wmma_gemm<2, false><<<g, 256, 0, stream>>>(A, fullRows, M, K, Bt, inp, msg, bo, a2m, sums);
  }
}

extern "C" void kernel_launch(void* const* d_in, const int* in_sizes, int n_in,
                              void* d_out, int out_size, void* d_ws, size_t ws_size,
                              hipStream_t stream) {
  const float* f_atoms = (const float*)d_in[0];
  const float* f_bonds = (const float*)d_in[1];
  const float* W_i     = (const float*)d_in[2];
  const float* W_h     = (const float*)d_in[3];
  const float* W_o     = (const float*)d_in[4];
  const float* b_o     = (const float*)d_in[5];
  const int*   a2b     = (const int*)d_in[6];
  const int*   b2a     = (const int*)d_in[7];
  const int*   b2revb  = (const int*)d_in[8];
  const int*   a2m     = (const int*)d_in[9];
  float* out = (float*)d_out;

  // carve workspace (256B aligned slabs)
  char* ws = (char*)d_ws;
  size_t off = 0;
  auto carve = [&](size_t bytes) -> void* {
    void* p = ws + off;
    off = (off + bytes + 255) & ~(size_t)255;
    return p;
  };
  float*          inp   = (float*)carve((size_t)N_BONDS * HIDDEN * 4);
  unsigned short* msg   = (unsigned short*)carve((size_t)N_BONDS * HIDDEN * 2);
  unsigned short* tbuf  = (unsigned short*)carve((size_t)N_BONDS * HIDDEN * 2);
  float*          amsg  = (float*)carve((size_t)N_ATOMS * HIDDEN * 4);
  unsigned short* fb    = (unsigned short*)carve((size_t)N_BONDS * KB_BONDS * 2);
  unsigned short* ain   = (unsigned short*)carve((size_t)N_ATOMS * KB_CAT * 2);
  unsigned short* WiT   = (unsigned short*)carve((size_t)HIDDEN * KB_BONDS * 2);
  unsigned short* WhT   = (unsigned short*)carve((size_t)HIDDEN * HIDDEN * 2);
  unsigned short* WoT   = (unsigned short*)carve((size_t)HIDDEN * KB_CAT * 2);
  float*          cnts  = (float*)carve((size_t)N_MOLS * 4);

  const int T = 256;
  // init output sums + counts every call (graph-replay safe)
  zero_f32<<<(N_MOLS * HIDDEN + T - 1) / T, T, 0, stream>>>(out, N_MOLS * HIDDEN);
  zero_f32<<<(N_MOLS + T - 1) / T, T, 0, stream>>>(cnts, N_MOLS);
  count_atoms<<<(N_ATOMS + T - 1) / T, T, 0, stream>>>(a2m, cnts);

  // weight conversion (bf16, transposed to N-major), bond-feature pad+convert
  conv_transpose<<<(HIDDEN * KB_BONDS + T - 1) / T, T, 0, stream>>>(W_i, WiT, BOND_FDIM, HIDDEN, KB_BONDS);
  conv_transpose<<<(HIDDEN * HIDDEN + T - 1) / T, T, 0, stream>>>(W_h, WhT, HIDDEN, HIDDEN, HIDDEN);
  conv_transpose<<<(HIDDEN * KB_CAT + T - 1) / T, T, 0, stream>>>(W_o, WoT, ATOM_FDIM + HIDDEN, HIDDEN, KB_CAT);
  conv_pad<<<((N_BONDS * KB_BONDS) + T - 1) / T, T, 0, stream>>>(f_bonds, fb, N_BONDS, BOND_FDIM, KB_BONDS);

  // inp = f_bonds @ W_i ; message = relu(inp)
  launch_gemm(0, fb, N_BONDS, KB_BONDS, WiT, inp, msg, nullptr, nullptr, nullptr, stream);

  // depth loop
  for (int d = 0; d < 4; ++d) {
    gather_atoms<<<N_ATOMS / 4, T, 0, stream>>>(msg, a2b, amsg);
    bond_pre<<<N_BONDS / 4, T, 0, stream>>>(amsg, msg, b2a, b2revb, tbuf);
    launch_gemm(1, tbuf, N_BONDS, HIDDEN, WhT, inp, msg, nullptr, nullptr, nullptr, stream);
  }

  // final gather + concat, readout GEMM with fused atomic segment-sum
  build_a_input<<<N_ATOMS, T, 0, stream>>>(f_atoms, msg, a2b, ain);
  launch_gemm(2, ain, N_ATOMS, KB_CAT, WoT, nullptr, nullptr, b_o, a2m, out, stream);

  // segment mean
  finalize_mean<<<(N_MOLS * HIDDEN + T - 1) / T, T, 0, stream>>>(out, cnts);
}
